// sLSTM_block_52733608460769
// MI455X (gfx1250) — compile-verified, matrix operand order checked by reading, not run on hardware
//
#include <hip/hip_runtime.h>
#include <hip/hip_bf16.h>

typedef __attribute__((ext_vector_type(16))) _Float16 v16h;
typedef __attribute__((ext_vector_type(8)))  _Float16 v8h;
typedef __attribute__((ext_vector_type(8)))  float    v8f;
typedef __attribute__((ext_vector_type(4)))  unsigned int u32x4;
typedef __attribute__((ext_vector_type(8)))  int          i32x8;
typedef __attribute__((ext_vector_type(4)))  int          i32x4;

#define DIMN  768
#define BB    4
#define LLEN  64
#define NHEAD 12
#define HDIM  64
#define HID   768
#define G4    3072           // 4*H gates
#define KCAT  1536           // D + H concat
#define BLD   (BB*LLEN*DIMN) // 196608

#if defined(__has_builtin)
#if __has_builtin(__builtin_amdgcn_tensor_load_to_lds) && __has_builtin(__builtin_amdgcn_s_wait_tensorcnt)
#define HAVE_TDM 1
#endif
#endif

// ---------------------------------------------------------------------------
// WMMA fragment loader: 16x32 f16 tile, row-major source with leading dim `ld`.
// CDNA5 wave32 A-layout: lanes 0-15 -> row m, K={0..7,16..23}; lanes 16-31 ->
// row m, K={8..15,24..31}. B (K x 16) uses the same pattern applied to rows of
// W (columns of B), so this loader serves both operands.
// ---------------------------------------------------------------------------
__device__ __forceinline__ v16h load_frag(const _Float16* __restrict__ base, int ld, int k0) {
  const int lane = threadIdx.x & 31;
  const int row  = lane & 15;
  const int koff = (lane < 16) ? 0 : 8;
  const _Float16* p = base + (size_t)row * ld + k0 + koff;
  v8h lo = *(const v8h*)(p);
  v8h hi = *(const v8h*)(p + 16);
  v16h r;
#pragma unroll
  for (int i = 0; i < 8; ++i) { r[i] = lo[i]; r[i + 8] = hi[i]; }
  return r;
}

__device__ __forceinline__ v8f wmma_f16(v16h a, v16h b, v8f c) {
  return __builtin_amdgcn_wmma_f32_16x16x32_f16(false, a, false, b, (short)0, c, false, false);
}

// ---------------------------------------------------------------------------
// Generic WMMA GEMM: each wave computes a 16(M) x 64(N) strip = 4 independent
// accumulator chains sharing one A fragment per K-step.
// C = scale * (A @ W^T) + bias (+C if accumulate) (+resid); optional f16 copy.
// zmode==1: per-head attention offsets (z = b*NH + h), S = q @ k^T.
// ---------------------------------------------------------------------------
__global__ void wmma_gemm_kernel(
    const _Float16* __restrict__ A, int lda,
    const _Float16* __restrict__ W, int ldw,
    const float* __restrict__ bias,
    const float* __restrict__ resid,
    float* __restrict__ C, _Float16* __restrict__ Ch, int ldc,
    int K, float scale, int accumulate, int zmode)
{
  long aOff, wOff, cOff;
  if (zmode == 1) {
    const int z = blockIdx.z, b = z / NHEAD, h = z % NHEAD;
    aOff = (long)b * (LLEN * DIMN) + h * HDIM;
    wOff = aOff;
    cOff = (long)z * (LLEN * LLEN);
  } else {
    aOff = 0; wOff = 0; cOff = 0;
  }
  const int n0base = blockIdx.x * 64;
  const int m0     = blockIdx.y * 16;
  const _Float16* Ab = A + aOff + (size_t)m0 * lda;
  const _Float16* W0 = W + wOff + (size_t)n0base * ldw;
  const int lane = threadIdx.x & 31;

  v8f acc[4];
#pragma unroll
  for (int i = 0; i < 4; ++i) acc[i] = (v8f){};

  for (int k0 = 0; k0 < K; k0 += 32) {
    __builtin_prefetch((const void*)(W0 + (size_t)(lane & 15) * ldw + k0 + 64), 0, 1);
    v16h a = load_frag(Ab, lda, k0);
#pragma unroll
    for (int i = 0; i < 4; ++i) {
      v16h b = load_frag(W0 + (size_t)(i * 16) * ldw, ldw, k0);
      acc[i] = wmma_f16(a, b, acc[i]);
    }
  }
  const int rbase = m0 + ((lane < 16) ? 0 : 8);
#pragma unroll
  for (int i = 0; i < 4; ++i) {
    const int col = n0base + i * 16 + (lane & 15);
    const float bvv = bias ? bias[col] : 0.f;
#pragma unroll
    for (int e = 0; e < 8; ++e) {
      const long off = cOff + (long)(rbase + e) * ldc + col;
      float val = acc[i][e] * scale + bvv;
      if (accumulate) val += C[off];
      if (resid)      val += resid[off];
      C[off] = val;
      if (Ch) Ch[off] = (_Float16)val;
    }
  }
}

// ---------------------------------------------------------------------------
// LayerNorm: one 256-thread block per row of 768; wave32 shuffle + LDS reduce.
// ---------------------------------------------------------------------------
__global__ void ln_kernel(const float* __restrict__ x, const float* __restrict__ g,
                          const float* __restrict__ bt, float* __restrict__ xn,
                          _Float16* __restrict__ xnh)
{
  const int row = blockIdx.x;
  const int t   = threadIdx.x;
  const float* xr = x + (size_t)row * DIMN;
  float v0 = xr[t], v1 = xr[t + 256], v2 = xr[t + 512];
  float s  = v0 + v1 + v2;
  float sq = v0 * v0 + v1 * v1 + v2 * v2;
#pragma unroll
  for (int o = 16; o > 0; o >>= 1) { s += __shfl_xor(s, o); sq += __shfl_xor(sq, o); }
  __shared__ float rs[8], rq[8], mv[2];
  const int wave = t >> 5, lane = t & 31;
  if (lane == 0) { rs[wave] = s; rq[wave] = sq; }
  __syncthreads();
  if (t == 0) {
    float S = 0.f, Q = 0.f;
    for (int i = 0; i < 8; ++i) { S += rs[i]; Q += rq[i]; }
    const float m = S / DIMN;
    mv[0] = m; mv[1] = rsqrtf(Q / DIMN - m * m + 1e-5f);
  }
  __syncthreads();
  const float m = mv[0], r = mv[1];
#pragma unroll
  for (int u = 0; u < 3; ++u) {
    const int c = t + u * 256;
    const float val = (xr[c] - m) * r * g[c] + bt[c];
    xn[(size_t)row * DIMN + c]  = val;
    xnh[(size_t)row * DIMN + c] = (_Float16)val;
  }
}

__global__ void cvt_f16_kernel(const float* __restrict__ src, _Float16* __restrict__ dst, int n) {
  const int i = blockIdx.x * blockDim.x + threadIdx.x;
  if (i < n) dst[i] = (_Float16)src[i];
}

// pack [Wih | Whh] -> f16 [3072, 1536]
__global__ void pack_lstm_w_kernel(const float* __restrict__ Wih, const float* __restrict__ Whh,
                                   _Float16* __restrict__ Wcat) {
  const int i = blockIdx.x * blockDim.x + threadIdx.x;
  if (i >= G4 * KCAT) return;
  const int n = i / KCAT, kk = i % KCAT;
  const float v = (kk < HID) ? Wih[n * HID + kk] : Whh[n * HID + (kk - HID)];
  Wcat[i] = (_Float16)v;
}

// conv_w [O,I,K] -> 3 f16 matrices [K][O*I]
__global__ void pack_conv_w_kernel(const float* __restrict__ cw, _Float16* __restrict__ dst) {
  const int i = blockIdx.x * blockDim.x + threadIdx.x;
  if (i >= DIMN * DIMN * 3) return;
  const int oi = i / 3, k = i % 3;
  dst[(size_t)k * (DIMN * DIMN) + oi] = (_Float16)cw[i];
}

// softmax over rows of 64, returning the row-sum of the softmax (faithful)
__global__ void softmax_rowsum_kernel(const float* __restrict__ S, float* __restrict__ rowsum) {
  const int r = blockIdx.x;
  const float* row = S + (size_t)r * LLEN;
  const int t = threadIdx.x;                     // 32 lanes, 2 elems each
  const float a = row[t], b = row[t + 32];
  float m = fmaxf(a, b);
#pragma unroll
  for (int o = 16; o > 0; o >>= 1) m = fmaxf(m, __shfl_xor(m, o));
  const float e = expf(a - m) + expf(b - m);
  float z = e;
#pragma unroll
  for (int o = 16; o > 0; o >>= 1) z += __shfl_xor(z, o);
  float u = e / z;
#pragma unroll
  for (int o = 16; o > 0; o >>= 1) u += __shfl_xor(u, o);
  if (t == 0) rowsum[r] = u;
}

__global__ void ctx_scale_kernel(const float* __restrict__ v, const float* __restrict__ rowsum,
                                 _Float16* __restrict__ ctxh) {
  const int i = blockIdx.x * blockDim.x + threadIdx.x;
  if (i >= BLD) return;
  const int b   = i / (LLEN * DIMN);
  const int rem = i % (LLEN * DIMN);
  const int l   = rem / DIMN;
  const int h   = (rem % DIMN) / HDIM;
  ctxh[i] = (_Float16)(v[i] * rowsum[(b * NHEAD + h) * LLEN + l]);
}

// 3 zero-padded shifted f16 copies of lstm output for K=3 'same' conv-as-GEMM
__global__ void shift_pack_kernel(const float* __restrict__ src, _Float16* __restrict__ dst) {
  const int i = blockIdx.x * blockDim.x + threadIdx.x;
  if (i >= 3 * BLD) return;
  const int k = i / BLD, rem = i % BLD;
  const int b = rem / (LLEN * DIMN);
  const int r2 = rem % (LLEN * DIMN);
  const int l = r2 / DIMN, c = r2 % DIMN;
  const int ts = l + k - 1;
  const float v = (ts >= 0 && ts < LLEN) ? src[((size_t)b * LLEN + ts) * DIMN + c] : 0.f;
  dst[i] = (_Float16)v;
}

// ---------------------------------------------------------------------------
// Persistent 2-layer xLSTM. One workgroup (16 waves): A = [x_t | h] in LDS
// (rows 4..15 zero-padded to M=16). Gate GEMM [16x1536]x[1536x3072]:
// k-outer loop, one LDS A-fragment shared by 12 independent accumulator
// chains per wave -> back-to-back independent v_wmma issue (no RAW stalls).
// Cell state lives in registers; h lives in the LDS A buffers as f16.
// ---------------------------------------------------------------------------
__device__ __forceinline__ float sigf(float x) { return 1.f / (1.f + expf(-x)); }

__device__ void lstm_step(const _Float16* __restrict__ A,
                          const _Float16* __restrict__ Wcat,
                          const float* __restrict__ Bsum,   // bih+bhh, in LDS
                          float* __restrict__ G, float* __restrict__ creg,
                          _Float16* __restrict__ hSelf, _Float16* __restrict__ yNext,
                          float* __restrict__ yOut, int tid)
{
  const int wave = tid >> 5, lane = tid & 31;
  v8f acc[12];
#pragma unroll
  for (int i = 0; i < 12; ++i) acc[i] = (v8f){};

  const _Float16* Wb = Wcat + (size_t)(wave * 16) * KCAT;   // tile i at +i*16*16*KCAT
  for (int k0 = 0; k0 < KCAT; k0 += 32) {
    v16h a = load_frag(A, KCAT, k0);
#pragma unroll
    for (int i = 0; i < 12; ++i) {
      v16h b = load_frag(Wb + (size_t)(i * 16 * 16) * KCAT, KCAT, k0);
      acc[i] = wmma_f16(a, b, acc[i]);
    }
  }
  if (lane < 16) {                    // rows 0..3 are the real batch rows
#pragma unroll
    for (int i = 0; i < 12; ++i) {
      const int col = (wave + i * 16) * 16 + lane;
      const float bvv = Bsum[col];
#pragma unroll
      for (int e = 0; e < 4; ++e) G[e * G4 + col] = acc[i][e] + bvv;
    }
  }
  __syncthreads();
#pragma unroll
  for (int u = 0; u < 6; ++u) {
    const int idx = tid + u * 512;    // 0..3071 over [4][768]
    const int b = idx / HID, j = idx % HID;
    const float gi = G[b * G4 + j];
    const float gf = G[b * G4 + HID + j];
    const float gg = G[b * G4 + 2 * HID + j];
    const float go = G[b * G4 + 3 * HID + j];
    const float c = sigf(gf) * creg[u] + sigf(gi) * tanhf(gg);
    creg[u] = c;
    const float h = sigf(go) * tanhf(c);
    hSelf[b * KCAT + j] = (_Float16)h;
    if (yNext) yNext[b * KCAT + j] = (_Float16)h;
    if (yOut)  yOut[(size_t)b * (LLEN * DIMN) + j] = h;
  }
  __syncthreads();
}

__global__ void __launch_bounds__(512) xlstm_kernel(
    const _Float16* __restrict__ ctxh,
    const _Float16* __restrict__ W0, const _Float16* __restrict__ W1,
    const float* __restrict__ bih0, const float* __restrict__ bhh0,
    const float* __restrict__ bih1, const float* __restrict__ bhh1,
    float* __restrict__ out)
{
  extern __shared__ char smem[];
  _Float16* A0  = (_Float16*)smem;              // [16][1536] = [x_t | h0]
  _Float16* A1  = A0 + 16 * KCAT;               // [16][1536] = [y0  | h1]
  float*    G   = (float*)(A1 + 16 * KCAT);     // [4][3072] gates
  float*    Bs0 = G + 4 * G4;                   // [3072] bih0+bhh0
  float*    Bs1 = Bs0 + G4;                     // [3072] bih1+bhh1
  const int tid = threadIdx.x;
  for (int i = tid; i < 16 * KCAT; i += 512) { A0[i] = (_Float16)0.f; A1[i] = (_Float16)0.f; }
  for (int i = tid; i < G4; i += 512) {
    Bs0[i] = bih0[i] + bhh0[i];
    Bs1[i] = bih1[i] + bhh1[i];
  }
  float c0r[6], c1r[6];
#pragma unroll
  for (int u = 0; u < 6; ++u) { c0r[u] = 0.f; c1r[u] = 0.f; }
  __syncthreads();

  for (int outer = 0; outer < LLEN; ++outer) {
    for (int t = 0; t < LLEN; ++t) {
#ifdef HAVE_TDM
      // TDM warm-up of the x_t rows into LDS: one descriptor per batch row
      // (1-D tile of 768 f16). D# packing per CDNA5 ISA.
      // 6-arg builtin (this toolchain): (g0 u32x4, g1 i32x8, i32x4, i32x4, i32x8, cpol)
      if (tid < 32) {
        const unsigned ldsBase = (unsigned)(size_t)(const void*)A0;
        const unsigned long long gBase =
            (unsigned long long)(size_t)(const void*)ctxh + (size_t)t * DIMN * 2;
#pragma unroll
        for (int b = 0; b < BB; ++b) {
          const unsigned long long ga = gBase + (unsigned long long)b * ((size_t)LLEN * DIMN * 2);
          u32x4 g0;
          g0[0] = 1u;                                        // count=1, user mode
          g0[1] = ldsBase + (unsigned)(b * KCAT * 2);        // lds_addr (bytes)
          g0[2] = (unsigned)(ga & 0xffffffffu);              // global_addr[31:0]
          g0[3] = (unsigned)((ga >> 32) & 0x1ffffffu) | (2u << 30);  // [56:32] | type=2
          i32x8 g1 = {0, 0, 0, 0, 0, 0, 0, 0};
          g1[0] = (1 << 16);                 // data_size=1 (2 bytes), wg_mask=0
          g1[1] = (DIMN & 0xffff) << 16;     // tensor_dim0[15:0] = 768
          g1[2] = (1 << 16);                 // tensor_dim1 = 1
          g1[3] = (DIMN & 0xffff) << 16;     // tile_dim0 = 768
          g1[5] = DIMN;                      // tensor_dim0_stride (unused for 1-D tile)
          i32x4 z4 = {0, 0, 0, 0};
          i32x8 z8 = {0, 0, 0, 0, 0, 0, 0, 0};
          __builtin_amdgcn_tensor_load_to_lds(g0, g1, z4, z4, z8, 0);
        }
        __builtin_amdgcn_s_wait_tensorcnt(0);
      }
#endif
      // authoritative copy (keeps semantics independent of TDM descriptor details)
      for (int i = tid; i < BB * DIMN; i += 512) {
        const int b = i / DIMN, c = i % DIMN;
        A0[b * KCAT + c] = ctxh[((size_t)b * LLEN + t) * DIMN + c];
      }
      __syncthreads();
      lstm_step(A0, W0, Bs0, G, c0r, A0 + HID, A1, nullptr, tid);
      lstm_step(A1, W1, Bs1, G, c1r, A1 + HID, nullptr, out + (size_t)t * DIMN, tid);
    }
  }
}

// ---------------------------------------------------------------------------
extern "C" void kernel_launch(void* const* d_in, const int* in_sizes, int n_in,
                              void* d_out, int out_size, void* d_ws, size_t ws_size,
                              hipStream_t stream) {
  (void)in_sizes; (void)n_in; (void)out_size; (void)ws_size;
  const float* x    = (const float*)d_in[0];
  const float* ln_g = (const float*)d_in[1];
  const float* ln_b = (const float*)d_in[2];
  const float* Wq   = (const float*)d_in[3];  const float* bq   = (const float*)d_in[4];
  const float* Wk   = (const float*)d_in[5];  const float* bk   = (const float*)d_in[6];
  const float* Wv   = (const float*)d_in[7];  const float* bv   = (const float*)d_in[8];
  const float* Wih0 = (const float*)d_in[9];  const float* Whh0 = (const float*)d_in[10];
  const float* bih0 = (const float*)d_in[11]; const float* bhh0 = (const float*)d_in[12];
  const float* Wih1 = (const float*)d_in[13]; const float* Whh1 = (const float*)d_in[14];
  const float* bih1 = (const float*)d_in[15]; const float* bhh1 = (const float*)d_in[16];
  const float* cw   = (const float*)d_in[17]; const float* cb   = (const float*)d_in[18];
  const float* Wssm = (const float*)d_in[19]; const float* bssm = (const float*)d_in[20];
  const float* Wout = (const float*)d_in[21]; const float* bout = (const float*)d_in[22];
  float* out = (float*)d_out;

  char* cur = (char*)d_ws;
  auto alloc = [&](size_t bytes) -> void* {
    void* p = (void*)cur;
    cur += (bytes + 255) & ~(size_t)255;
    return p;
  };
  float*    xn    = (float*)alloc((size_t)BLD * 4);
  _Float16* xnh   = (_Float16*)alloc((size_t)BLD * 2);
  float*    qf    = (float*)alloc((size_t)BLD * 4);
  float*    kfb   = (float*)alloc((size_t)BLD * 4);
  float*    vf    = (float*)alloc((size_t)BLD * 4);
  _Float16* qh    = (_Float16*)alloc((size_t)BLD * 2);
  _Float16* kh    = (_Float16*)alloc((size_t)BLD * 2);
  float*    S     = (float*)alloc((size_t)BB * NHEAD * LLEN * LLEN * 4);
  float*    rsum  = (float*)alloc((size_t)BB * NHEAD * LLEN * 4);
  _Float16* ctxh  = (_Float16*)alloc((size_t)BLD * 2);
  _Float16* Wqh   = (_Float16*)alloc((size_t)DIMN * DIMN * 2);
  _Float16* Wkh   = (_Float16*)alloc((size_t)DIMN * DIMN * 2);
  _Float16* Wvh   = (_Float16*)alloc((size_t)DIMN * DIMN * 2);
  _Float16* Wssmh = (_Float16*)alloc((size_t)DIMN * DIMN * 2);
  _Float16* Wouth = (_Float16*)alloc((size_t)DIMN * DIMN * 2);
  _Float16* W0h   = (_Float16*)alloc((size_t)G4 * KCAT * 2);
  _Float16* W1h   = (_Float16*)alloc((size_t)G4 * KCAT * 2);
  _Float16* Wch   = (_Float16*)alloc((size_t)3 * DIMN * DIMN * 2);
  float*    lout  = (float*)alloc((size_t)BLD * 4);
  _Float16* shb   = (_Float16*)alloc((size_t)3 * BLD * 2);
  float*    cout  = (float*)alloc((size_t)BLD * 4);
  _Float16* couth = (_Float16*)alloc((size_t)BLD * 2);
  float*    y1    = (float*)alloc((size_t)BLD * 4);
  _Float16* y1h   = (_Float16*)alloc((size_t)BLD * 2);

  auto grid1 = [](int n) { return dim3((unsigned)((n + 255) / 256)); };

  ln_kernel<<<BB * LLEN, 256, 0, stream>>>(x, ln_g, ln_b, xn, xnh);

  cvt_f16_kernel<<<grid1(DIMN * DIMN), 256, 0, stream>>>(Wq, Wqh, DIMN * DIMN);
  cvt_f16_kernel<<<grid1(DIMN * DIMN), 256, 0, stream>>>(Wk, Wkh, DIMN * DIMN);
  cvt_f16_kernel<<<grid1(DIMN * DIMN), 256, 0, stream>>>(Wv, Wvh, DIMN * DIMN);
  cvt_f16_kernel<<<grid1(DIMN * DIMN), 256, 0, stream>>>(Wssm, Wssmh, DIMN * DIMN);
  cvt_f16_kernel<<<grid1(DIMN * DIMN), 256, 0, stream>>>(Wout, Wouth, DIMN * DIMN);
  pack_lstm_w_kernel<<<grid1(G4 * KCAT), 256, 0, stream>>>(Wih0, Whh0, W0h);
  pack_lstm_w_kernel<<<grid1(G4 * KCAT), 256, 0, stream>>>(Wih1, Whh1, W1h);
  pack_conv_w_kernel<<<grid1(DIMN * DIMN * 3), 256, 0, stream>>>(cw, Wch);

  // q,k,v projections: [256,768] x [768,768]^T  (16x64 strip per wave)
  dim3 gProj(DIMN / 64, (BB * LLEN) / 16, 1);
  wmma_gemm_kernel<<<gProj, 32, 0, stream>>>(xnh, DIMN, Wqh, DIMN, bq, nullptr, qf, qh, DIMN,
                                             DIMN, 1.f, 0, 0);
  wmma_gemm_kernel<<<gProj, 32, 0, stream>>>(xnh, DIMN, Wkh, DIMN, bk, nullptr, kfb, kh, DIMN,
                                             DIMN, 1.f, 0, 0);
  wmma_gemm_kernel<<<gProj, 32, 0, stream>>>(xnh, DIMN, Wvh, DIMN, bv, nullptr, vf, nullptr, DIMN,
                                             DIMN, 1.f, 0, 0);

  // attention scores per (b,h): S = q @ k^T / sqrt(D)
  dim3 gAttn(LLEN / 64, LLEN / 16, BB * NHEAD);
  wmma_gemm_kernel<<<gAttn, 32, 0, stream>>>(qh, DIMN, kh, DIMN, nullptr, nullptr, S, nullptr,
                                             LLEN, HDIM, 1.f / sqrtf((float)DIMN), 0, 1);
  softmax_rowsum_kernel<<<BB * NHEAD * LLEN, 32, 0, stream>>>(S, rsum);
  ctx_scale_kernel<<<grid1(BLD), 256, 0, stream>>>(vf, rsum, ctxh);

  // persistent recurrent core: 64 outer x 64 timesteps x 2 layers
  const size_t lds = (size_t)2 * 16 * KCAT * 2   // A0, A1
                   + (size_t)4 * G4 * 4          // gates
                   + (size_t)2 * G4 * 4;         // bias sums  (168 KB total)
  xlstm_kernel<<<1, 512, lds, stream>>>(ctxh, W0h, W1h, bih0, bhh0, bih1, bhh1, lout);

  // conv1d 'same' K=3 as 3 accumulating GEMMs over shifted copies
  shift_pack_kernel<<<grid1(3 * BLD), 256, 0, stream>>>(lout, shb);
  for (int k = 0; k < 3; ++k) {
    wmma_gemm_kernel<<<gProj, 32, 0, stream>>>(shb + (size_t)k * BLD, DIMN,
                                               Wch + (size_t)k * DIMN * DIMN, DIMN,
                                               (k == 0) ? cb : nullptr, nullptr,
                                               cout, (k == 2) ? couth : nullptr, DIMN,
                                               DIMN, 1.f, (k > 0) ? 1 : 0, 0);
  }

  // y @ Wssm^T + bssm ; then @ Wout^T + bout + xn (residual fused in epilogue)
  wmma_gemm_kernel<<<gProj, 32, 0, stream>>>(couth, DIMN, Wssmh, DIMN, bssm, nullptr, y1, y1h,
                                             DIMN, DIMN, 1.f, 0, 0);
  wmma_gemm_kernel<<<gProj, 32, 0, stream>>>(y1h, DIMN, Wouth, DIMN, bout, xn, out, nullptr,
                                             DIMN, DIMN, 1.f, 0, 0);
}
